// WindowMultiHeadAttention_76467597738325
// MI455X (gfx1250) — matrix operand order repaired and checked by reading, not compiled, not run on hardware
//
#include <hip/hip_runtime.h>
#include <hip/hip_bf16.h>

// ---------------------------------------------------------------------------
// Swin window attention for MI455X (gfx1250, wave32).
// GEMMs: v_wmma_f32_16x16x32_bf16 (f32 accumulate), LDS-staged tiles.
// B-tiles (and proj A-tiles) staged by the Tensor Data Mover.
// Workspace layout (bytes):
//   [0)          wqkvT  bf16 [1536][512]   1,572,864
//   [1572864)    wprojT bf16 [512][512]      524,288
//   [2097152)    qkv    bf16 [100352][1536] 308,281,344
//   [310378496)  attnb  bf16 [100352][512]  102,760,448
// ---------------------------------------------------------------------------

typedef __bf16 bf16_t;
typedef bf16_t v16bf __attribute__((ext_vector_type(16)));
typedef bf16_t v2bf  __attribute__((ext_vector_type(2)));
typedef float  v8f   __attribute__((ext_vector_type(8)));
typedef unsigned int u32x4 __attribute__((ext_vector_type(4)));
typedef int    i32x8 __attribute__((ext_vector_type(8)));
typedef int    i32x4 __attribute__((ext_vector_type(4)));
typedef float  f32x4 __attribute__((ext_vector_type(4)));

#define LTOK   49
#define NHEAD  16
#define HDIM   32
#define CDIM   512
#define NQKV   1536
#define MROWS  100352      // 2048 windows * 49 tokens
#define SCALE_F 0.17677669529663687f   // 32^-0.5

static __device__ __forceinline__ bf16_t tobf(float f) { return (bf16_t)f; }

// A-fragment (16x32 bf16, lane=M%16): 16 contiguous K halves starting at g=0|16
static __device__ __forceinline__ v16bf ldsA(const bf16_t* p) {
  union { v16bf v; u32x4 u[2]; } t;
  t.u[0] = *(const u32x4*)(p);
  t.u[1] = *(const u32x4*)(p + 8);
  return t.v;
}
// B-fragment (32x16 bf16, lane=N%16): 16 contiguous K halves starting at g=0|16
static __device__ __forceinline__ v16bf ldsB(const bf16_t* p) {
  union { v16bf v; u32x4 u[2]; } t;
  t.u[0] = *(const u32x4*)(p);
  t.u[1] = *(const u32x4*)(p + 8);
  return t.v;
}

#define WMMA_BF16(A, B, C) \
  __builtin_amdgcn_wmma_f32_16x16x32_bf16(false, (A), false, (B), (short)0, (C), false, false)

#define HAVE_TDM __has_builtin(__builtin_amdgcn_tensor_load_to_lds)

#if HAVE_TDM
// Tensor Data Mover: 2D tile (tileX x tileY elems, 2B/elem) global -> LDS.
// pad_enable inserts 4 DWORDs (8 halves) after every 32 DWORDs (64 halves):
// with tileX == 64 this yields LDS row stride 72 halves (matches ldsA/ldsB).
static __device__ __forceinline__ void tdm_load_2d(
    unsigned ldsOff, const void* gptr,
    unsigned tileX, unsigned tileY,
    unsigned tensorX, unsigned tensorY, unsigned rowStride)
{
  unsigned long long ga = (unsigned long long)gptr;
  u32x4 g0;
  g0[0] = 1u;                                            // count=1, user mode
  g0[1] = ldsOff;                                        // lds_addr (bytes)
  g0[2] = (unsigned)(ga & 0xFFFFFFFFu);                  // global_addr[31:0]
  g0[3] = (unsigned)((ga >> 32) & 0x1FFFFFFu) | (2u << 30);  // [56:32], type=2
  i32x8 g1;
  g1[0] = (1 << 16) | (1 << 20) | (4 << 22) | (3 << 25); // 2B elems, pad 4dw/32dw
  g1[1] = (int)((tensorX & 0xFFFFu) << 16);              // tensor_dim0[15:0]
  g1[2] = (int)(((tensorX >> 16) & 0xFFFFu) | ((tensorY & 0xFFFFu) << 16));
  g1[3] = (int)(((tensorY >> 16) & 0xFFFFu) | (tileX << 16));   // tile_dim0
  g1[4] = (int)(tileY & 0xFFFFu);                        // tile_dim1, tile_dim2=0
  g1[5] = (int)rowStride;                                // tensor_dim0_stride
  g1[6] = 0;
  g1[7] = 0;
  i32x4 z4 = {0, 0, 0, 0};
  i32x8 z8 = {0, 0, 0, 0, 0, 0, 0, 0};
  __builtin_amdgcn_tensor_load_to_lds(g0, g1, z4, z4, z8, 0);
}
#endif

// ---------------------------------------------------------------------------
// K0: transpose + f32->bf16 weights:  wqkvT[n][k], wprojT[n][k]
// ---------------------------------------------------------------------------
__global__ __launch_bounds__(256) void wconv(
    const float* __restrict__ wqkv, const float* __restrict__ wproj,
    bf16_t* __restrict__ wqkvT, bf16_t* __restrict__ wprojT)
{
  int e = blockIdx.x * 256 + threadIdx.x;
  if (e < NQKV * CDIM) {
    int n = e / CDIM, k = e - n * CDIM;
    wqkvT[e] = tobf(wqkv[(size_t)k * NQKV + n]);
  }
  int e2 = e - NQKV * CDIM;
  if (e2 >= 0 && e2 < CDIM * CDIM) {
    int n = e2 / CDIM, k = e2 - n * CDIM;
    wprojT[e2] = tobf(wproj[(size_t)k * CDIM + n]);
  }
}

// ---------------------------------------------------------------------------
// K1: qkv = bf16( x[100352x512] @ w_qkv[512x1536] )
// 128x64 workgroup tile, K-chunks of 64; 8 waves x 4 accumulators (2M x 2N).
// A: clause-batched f32 loads + packed cvt -> LDS.  B: TDM DMA -> LDS.
// ---------------------------------------------------------------------------
__global__ __launch_bounds__(256) void qkv_gemm(
    const float* __restrict__ x, const bf16_t* __restrict__ wT,
    bf16_t* __restrict__ qkv)
{
  __shared__ bf16_t As[128 * 72];   // 18 KB
  __shared__ bf16_t Bs[64 * 72];    //  9 KB

  const int t = threadIdx.x;
  const int lane = t & 31, wv = t >> 5;
  const int row0 = blockIdx.x * 128;
  const int n0   = blockIdx.y * 64;
  const int mi0 = (wv & 3) * 2;     // M subtile pair base (0,2,4,6)
  const int ni0 = (wv >> 2) * 2;    // N subtile pair base (0,2)
  const int mloc = lane & 15;
  const int ncol = lane & 15;
  const int grpA = (lane < 16) ? 0 : 16;
  const int grpB = (lane < 16) ? 0 : 16;
#if HAVE_TDM
  const unsigned bsOff = (unsigned)(size_t)(void*)Bs;
#endif

  v8f acc00 = {}, acc01 = {}, acc10 = {}, acc11 = {};

  for (int kk = 0; kk < CDIM; kk += 64) {
#if HAVE_TDM
    if (wv == 0) {
      tdm_load_2d(bsOff, wT + (size_t)n0 * CDIM + kk, 64, 64,
                  CDIM, NQKV, CDIM);
    }
#else
#pragma unroll
    for (int i = 0; i < 2; ++i) {
      int f = t + i * 256;
      int r = f >> 3, c8 = (f & 7) << 3;
      *(u32x4*)(Bs + r * 72 + c8) =
          *(const u32x4*)(wT + (size_t)(n0 + r) * CDIM + kk + c8);
    }
#endif
    // stage A: 128x64 f32 -> bf16. Batch all 8 loads first (clause + pipelined
    // waits), then packed pair converts + 8B LDS stores.
    f32x4 av[8];
#pragma unroll
    for (int i = 0; i < 8; ++i) {
      int f = t + i * 256;                 // 0..2047 float4 units
      int r = f >> 4, c4 = (f & 15) << 2;
      av[i] = *(const f32x4*)(x + (size_t)(row0 + r) * CDIM + kk + c4);
    }
#pragma unroll
    for (int i = 0; i < 8; ++i) {
      int f = t + i * 256;
      int r = f >> 4, c4 = (f & 15) << 2;
      v2bf p0; p0[0] = tobf(av[i][0]); p0[1] = tobf(av[i][1]);
      v2bf p1; p1[0] = tobf(av[i][2]); p1[1] = tobf(av[i][3]);
      *(v2bf*)(As + r * 72 + c4)     = p0;
      *(v2bf*)(As + r * 72 + c4 + 2) = p1;
    }
#if HAVE_TDM
    if (wv == 0) __builtin_amdgcn_s_wait_tensorcnt(0);
#endif
    __syncthreads();

#pragma unroll
    for (int ks = 0; ks < 64; ks += 32) {
      v16bf a0 = ldsA(As + ((mi0 + 0) * 16 + mloc) * 72 + ks + grpA);
      v16bf a1 = ldsA(As + ((mi0 + 1) * 16 + mloc) * 72 + ks + grpA);
      v16bf b0 = ldsB(Bs + ((ni0 + 0) * 16 + ncol) * 72 + ks + grpB);
      v16bf b1 = ldsB(Bs + ((ni0 + 1) * 16 + ncol) * 72 + ks + grpB);
      acc00 = WMMA_BF16(a0, b0, acc00);
      acc01 = WMMA_BF16(a0, b1, acc01);
      acc10 = WMMA_BF16(a1, b0, acc10);
      acc11 = WMMA_BF16(a1, b1, acc11);
    }
    __syncthreads();
  }

  const int m0 = (lane < 16) ? 0 : 8;
#pragma unroll
  for (int r = 0; r < 8; ++r) {
    size_t g0 = (size_t)(row0 + (mi0 + 0) * 16 + m0 + r);
    size_t g1 = (size_t)(row0 + (mi0 + 1) * 16 + m0 + r);
    qkv[g0 * NQKV + n0 + (ni0 + 0) * 16 + ncol] = tobf(acc00[r]);
    qkv[g0 * NQKV + n0 + (ni0 + 1) * 16 + ncol] = tobf(acc01[r]);
    qkv[g1 * NQKV + n0 + (ni0 + 0) * 16 + ncol] = tobf(acc10[r]);
    qkv[g1 * NQKV + n0 + (ni0 + 1) * 16 + ncol] = tobf(acc11[r]);
  }
}

// ---------------------------------------------------------------------------
// K2: per (window, head) attention. 4 waves, each owns 16 query rows.
// ---------------------------------------------------------------------------
__global__ __launch_bounds__(128) void attn_kernel(
    const bf16_t* __restrict__ qkv, const float* __restrict__ mask,
    const float* __restrict__ bias_table, bf16_t* __restrict__ attnb)
{
  __shared__ bf16_t Qs[64 * 40];       // [token][d], rows >=49 zeroed
  __shared__ bf16_t Ks[64 * 40];       // [token][d]  (== B-layout of K^T)
  __shared__ bf16_t Vt[32 * 72];       // [d][token]  (B-layout of V)
  __shared__ bf16_t Ps[4][16 * 72];    // per-wave softmax probs, A-layout

  const int b = blockIdx.x;            // window 0..2047
  const int h = blockIdx.y;            // head 0..15
  const int t = threadIdx.x;
  const int lane = t & 31, wv = t >> 5;
  const size_t rowbase = (size_t)b * LTOK;

  // ---- stage Q, K: 2 threads per token, 32B each ----
  {
    int token = t >> 1;
    int d0 = (t & 1) * 16;
    if (token < LTOK) {
      const bf16_t* sq = qkv + (rowbase + token) * NQKV + h * HDIM + d0;
      *(u32x4*)(Qs + token * 40 + d0)     = *(const u32x4*)(sq);
      *(u32x4*)(Qs + token * 40 + d0 + 8) = *(const u32x4*)(sq + 8);
      const bf16_t* sk = sq + CDIM;
      *(u32x4*)(Ks + token * 40 + d0)     = *(const u32x4*)(sk);
      *(u32x4*)(Ks + token * 40 + d0 + 8) = *(const u32x4*)(sk + 8);
    } else {
      u32x4 z = {};
      *(u32x4*)(Qs + token * 40 + d0) = z; *(u32x4*)(Qs + token * 40 + d0 + 8) = z;
      *(u32x4*)(Ks + token * 40 + d0) = z; *(u32x4*)(Ks + token * 40 + d0 + 8) = z;
    }
  }
  // ---- stage V transposed: Vt[d][token] ----
#pragma unroll
  for (int i = 0; i < 16; ++i) {
    int e = t * 16 + i;                // 0..2047
    int d = e >> 6, tok = e & 63;
    bf16_t v = (tok < LTOK)
        ? qkv[(rowbase + tok) * NQKV + 2 * CDIM + h * HDIM + d] : tobf(0.0f);
    Vt[d * 72 + tok] = v;
  }
  __syncthreads();

  const int grpA = (lane < 16) ? 0 : 16;
  const int grpB = (lane < 16) ? 0 : 16;
  const int ncol = lane & 15;
  const int mloc = lane & 15;
  const int m0 = (lane < 16) ? 0 : 8;

  // ---- S = Q K^T : one wmma per 16-key block (K-dim = HD = 32) ----
  v8f s[4] = {{}, {}, {}, {}};
  v16bf aq = ldsA(Qs + (wv * 16 + mloc) * 40 + grpA);
#pragma unroll
  for (int nb = 0; nb < 4; ++nb) {
    v16bf bk = ldsB(Ks + (nb * 16 + ncol) * 40 + grpB);
    s[nb] = WMMA_BF16(aq, bk, s[nb]);
  }

  // ---- scale + relative-position bias + window mask ----
  const int wmask = b & 63;
  float val[4][8];
#pragma unroll
  for (int nb = 0; nb < 4; ++nb) {
    int kj = nb * 16 + ncol;
    int jh = kj / 7, jw = kj - jh * 7;
#pragma unroll
    for (int r = 0; r < 8; ++r) {
      int qi = wv * 16 + m0 + r;
      float v = s[nb][r] * SCALE_F;
      if (kj >= LTOK) {
        v = -3.0e30f;                          // padded keys -> 0 after softmax
      } else if (qi < LTOK) {
        int ih = qi / 7, iw = qi - ih * 7;
        int ridx = (ih - jh + 6) * 13 + (iw - jw + 6);
        v += bias_table[ridx * NHEAD + h]
           + mask[(size_t)wmask * (LTOK * LTOK) + qi * LTOK + kj];
      }
      val[nb][r] = v;
    }
  }

  // ---- softmax: rows live across 16 lanes at fixed accumulator index ----
#pragma unroll
  for (int r = 0; r < 8; ++r) {
    float m = fmaxf(fmaxf(val[0][r], val[1][r]), fmaxf(val[2][r], val[3][r]));
    for (int off = 8; off; off >>= 1) m = fmaxf(m, __shfl_xor(m, off, 16));
    float sum = 0.0f;
#pragma unroll
    for (int nb = 0; nb < 4; ++nb) { val[nb][r] = __expf(val[nb][r] - m); sum += val[nb][r]; }
    for (int off = 8; off; off >>= 1) sum += __shfl_xor(sum, off, 16);
    float inv = 1.0f / sum;
#pragma unroll
    for (int nb = 0; nb < 4; ++nb) val[nb][r] *= inv;
  }

  // ---- restage P into A-layout via wave-private LDS ----
#pragma unroll
  for (int nb = 0; nb < 4; ++nb)
#pragma unroll
    for (int r = 0; r < 8; ++r)
      Ps[wv][(m0 + r) * 72 + nb * 16 + ncol] = tobf(val[nb][r]);

  // ---- O = P @ V : 2 K-steps x 2 d-blocks ----
  v8f o[2] = {{}, {}};
#pragma unroll
  for (int ks = 0; ks < 64; ks += 32) {
    v16bf ap = ldsA(Ps[wv] + mloc * 72 + ks + grpA);
#pragma unroll
    for (int n2 = 0; n2 < 2; ++n2) {
      v16bf bv = ldsB(Vt + (n2 * 16 + ncol) * 72 + ks + grpB);
      o[n2] = WMMA_BF16(ap, bv, o[n2]);
    }
  }
#pragma unroll
  for (int n2 = 0; n2 < 2; ++n2)
#pragma unroll
    for (int r = 0; r < 8; ++r) {
      int qi = wv * 16 + m0 + r;
      if (qi < LTOK)
        attnb[(rowbase + qi) * CDIM + h * HDIM + n2 * 16 + ncol] = tobf(o[n2][r]);
    }
}

// ---------------------------------------------------------------------------
// K3: out = attnb[100352x512] @ w_proj[512x512]  -> f32
// 128x64 workgroup tile; 8 waves x 4 accumulators; A and B tiles via TDM.
// ---------------------------------------------------------------------------
__global__ __launch_bounds__(256) void proj_gemm(
    const bf16_t* __restrict__ ain, const bf16_t* __restrict__ wT,
    float* __restrict__ out)
{
  __shared__ bf16_t As[128 * 72];
  __shared__ bf16_t Bs[64 * 72];

  const int t = threadIdx.x;
  const int lane = t & 31, wv = t >> 5;
  const int row0 = blockIdx.x * 128;
  const int n0   = blockIdx.y * 64;
  const int mi0 = (wv & 3) * 2;
  const int ni0 = (wv >> 2) * 2;
  const int mloc = lane & 15, ncol = lane & 15;
  const int grpA = (lane < 16) ? 0 : 16;
  const int grpB = (lane < 16) ? 0 : 16;
#if HAVE_TDM
  const unsigned asOff = (unsigned)(size_t)(void*)As;
  const unsigned bsOff = (unsigned)(size_t)(void*)Bs;
#endif

  v8f acc00 = {}, acc01 = {}, acc10 = {}, acc11 = {};

  for (int kk = 0; kk < CDIM; kk += 64) {
#if HAVE_TDM
    if (wv == 0) {
      tdm_load_2d(asOff, ain + (size_t)row0 * CDIM + kk, 64, 128,
                  CDIM, MROWS, CDIM);
      tdm_load_2d(bsOff, wT + (size_t)n0 * CDIM + kk, 64, 64,
                  CDIM, CDIM, CDIM);
      __builtin_amdgcn_s_wait_tensorcnt(0);
    }
#else
#pragma unroll
    for (int i = 0; i < 4; ++i) {
      int f = t + i * 256;
      int r = f >> 3, c8 = (f & 7) << 3;
      *(u32x4*)(As + r * 72 + c8) =
          *(const u32x4*)(ain + (size_t)(row0 + r) * CDIM + kk + c8);
    }
#pragma unroll
    for (int i = 0; i < 2; ++i) {
      int f = t + i * 256;
      int r = f >> 3, c8 = (f & 7) << 3;
      *(u32x4*)(Bs + r * 72 + c8) =
          *(const u32x4*)(wT + (size_t)(n0 + r) * CDIM + kk + c8);
    }
#endif
    __syncthreads();

#pragma unroll
    for (int ks = 0; ks < 64; ks += 32) {
      v16bf a0 = ldsA(As + ((mi0 + 0) * 16 + mloc) * 72 + ks + grpA);
      v16bf a1 = ldsA(As + ((mi0 + 1) * 16 + mloc) * 72 + ks + grpA);
      v16bf b0 = ldsB(Bs + ((ni0 + 0) * 16 + ncol) * 72 + ks + grpB);
      v16bf b1 = ldsB(Bs + ((ni0 + 1) * 16 + ncol) * 72 + ks + grpB);
      acc00 = WMMA_BF16(a0, b0, acc00);
      acc01 = WMMA_BF16(a0, b1, acc01);
      acc10 = WMMA_BF16(a1, b0, acc10);
      acc11 = WMMA_BF16(a1, b1, acc11);
    }
    __syncthreads();
  }

  const int m0 = (lane < 16) ? 0 : 8;
#pragma unroll
  for (int r = 0; r < 8; ++r) {
    size_t g0 = (size_t)(row0 + (mi0 + 0) * 16 + m0 + r);
    size_t g1 = (size_t)(row0 + (mi0 + 1) * 16 + m0 + r);
    out[g0 * CDIM + n0 + (ni0 + 0) * 16 + ncol] = acc00[r];
    out[g0 * CDIM + n0 + (ni0 + 1) * 16 + ncol] = acc01[r];
    out[g1 * CDIM + n0 + (ni0 + 0) * 16 + ncol] = acc10[r];
    out[g1 * CDIM + n0 + (ni0 + 1) * 16 + ncol] = acc11[r];
  }
}

// ---------------------------------------------------------------------------
extern "C" void kernel_launch(void* const* d_in, const int* in_sizes, int n_in,
                              void* d_out, int out_size, void* d_ws, size_t ws_size,
                              hipStream_t stream) {
  const float* x          = (const float*)d_in[0];
  const float* mask       = (const float*)d_in[1];
  const float* bias_table = (const float*)d_in[2];
  const float* w_qkv      = (const float*)d_in[3];
  const float* w_proj     = (const float*)d_in[4];
  float* out = (float*)d_out;

  char* ws = (char*)d_ws;
  bf16_t* wqkvT  = (bf16_t*)(ws);
  bf16_t* wprojT = (bf16_t*)(ws + 1572864);
  bf16_t* qkv    = (bf16_t*)(ws + 2097152);
  bf16_t* attnb  = (bf16_t*)(ws + 2097152 + 308281344ULL);

  wconv<<<dim3(4096), dim3(256), 0, stream>>>(w_qkv, w_proj, wqkvT, wprojT);
  qkv_gemm<<<dim3(784, 24), dim3(256), 0, stream>>>(x, wqkvT, qkv);
  attn_kernel<<<dim3(2048, 16), dim3(128), 0, stream>>>(qkv, mask, bias_table, attnb);
  proj_gemm<<<dim3(784, 8), dim3(256), 0, stream>>>(attnb, wprojT, out);
}
